// GptOssExperts_19774029431102
// MI455X (gfx1250) — matrix operand order repaired and checked by reading, not compiled, not run on hardware
//
#include <hip/hip_runtime.h>
#include <hip/hip_bf16.h>

// Problem constants (match reference)
#define NE 16
#define NH 1024
#define ND 1024
#define NT 2048
#define ALPHA_ 1.702f
#define LIMIT_ 7.0f

typedef __attribute__((ext_vector_type(16))) __bf16       bf16x16;
typedef __attribute__((ext_vector_type(8)))  float        f32x8;
typedef __attribute__((ext_vector_type(4)))  unsigned int u32x4;

union FragU { bf16x16 v; u32x4 q[2]; };

// A-matrix 16x32 bf16 fragment (CDNA5 layout: lanes 0-15 hold K=[0..7]+[16..23],
// lanes 16-31 hold K=[8..15]+[24..31] of row M=lane&15). Two b128 loads.
__device__ __forceinline__ bf16x16 load_a_frag(const __bf16* rowBase, int ld, int kk, int lane) {
  const int r = lane & 15, half = lane >> 4;
  const __bf16* p = rowBase + (size_t)r * ld + kk + half * 8;
  FragU f;
  f.q[0] = *reinterpret_cast<const u32x4*>(p);
  f.q[1] = *reinterpret_cast<const u32x4*>(p + 16);
  return f.v;
}

// B-matrix 32x16 bf16 fragment (lanes hold column N=lane&15; lanes 0-15 carry
// K=[0..7]+[16..23], lanes 16-31 carry K=[8..15]+[24..31]). Two b128 loads.
__device__ __forceinline__ bf16x16 load_b_frag(const __bf16* rowBase, int ld, int kk, int lane) {
  const int n = lane & 15, half = lane >> 4;
  const __bf16* p = rowBase + (size_t)n * ld + kk + half * 8;
  FragU f;
  f.q[0] = *reinterpret_cast<const u32x4*>(p);
  f.q[1] = *reinterpret_cast<const u32x4*>(p + 16);
  return f.v;
}

#define WMMA_BF16(a, b, c) \
  __builtin_amdgcn_wmma_f32_16x16x32_bf16(false, (a), false, (b), (short)0, (c), false, false)

// ---------------------------------------------------------------------------
// K0: fp32 -> bf16 cast (grid-stride)
// ---------------------------------------------------------------------------
__global__ __launch_bounds__(256) void k_cast_bf16(const float* __restrict__ in,
                                                   __bf16* __restrict__ out, int n) {
  int i = blockIdx.x * blockDim.x + threadIdx.x;
  int stride = gridDim.x * blockDim.x;
  for (; i < n; i += stride) out[i] = (__bf16)in[i];
}

// ---------------------------------------------------------------------------
// K1: gate_up GEMM (bf16 WMMA, f32 acc) + bias + swiglu, fused -> bf16 [E,T,D]
// block = 256 thr (8 waves, 2x4); wave tile = 32 tokens x 64 gate_up cols
// block tile = 64 tokens x 256 gate_up cols; grid = (2D/256=8, T/64=32, E=16)
// Per K-step per wave: 2 A-frags + 4 B-frags (12 b128 loads) feed 8 WMMAs.
// ---------------------------------------------------------------------------
__global__ __launch_bounds__(256) void k_gateup_swiglu(
    const __bf16* __restrict__ xb,    // [T,H]
    const __bf16* __restrict__ wgu,   // [E,2D,H]
    const float*  __restrict__ bgu,   // [E,2D]
    __bf16* __restrict__ fused)       // [E,T,D]
{
  const int lane = threadIdx.x & 31;
  const int wave = threadIdx.x >> 5;
  const int wm = wave >> 2;                       // 0..1 (token dir)
  const int wn = wave & 3;                        // 0..3 (col dir)
  const int e  = blockIdx.z;
  const int cBase = blockIdx.x * 256 + wn * 64;   // gate_up column base (even)
  const int tBase = blockIdx.y * 64 + wm * 32;    // token base

  const __bf16* we = wgu + (size_t)e * (2 * ND) * NH;
  const __bf16* aRow0 = xb + (size_t)tBase * NH;
  const __bf16* aRow1 = aRow0 + (size_t)16 * NH;
  const __bf16* bRow[4] = {
    we + (size_t)cBase * NH,
    we + (size_t)(cBase + 16) * NH,
    we + (size_t)(cBase + 32) * NH,
    we + (size_t)(cBase + 48) * NH };

  f32x8 c[2][4] = {};
  for (int kk = 0; kk < NH; kk += 32) {
    bf16x16 a0 = load_a_frag(aRow0, NH, kk, lane);
    bf16x16 a1 = load_a_frag(aRow1, NH, kk, lane);
    bf16x16 b0 = load_b_frag(bRow[0], NH, kk, lane);
    bf16x16 b1 = load_b_frag(bRow[1], NH, kk, lane);
    bf16x16 b2 = load_b_frag(bRow[2], NH, kk, lane);
    bf16x16 b3 = load_b_frag(bRow[3], NH, kk, lane);
    // prefetch next K slab of the weight stream into caches
    __builtin_prefetch((const void*)(bRow[0] + (size_t)(lane & 15) * NH + kk + 32), 0, 1);
    c[0][0] = WMMA_BF16(a0, b0, c[0][0]);
    c[0][1] = WMMA_BF16(a0, b1, c[0][1]);
    c[0][2] = WMMA_BF16(a0, b2, c[0][2]);
    c[0][3] = WMMA_BF16(a0, b3, c[0][3]);
    c[1][0] = WMMA_BF16(a1, b0, c[1][0]);
    c[1][1] = WMMA_BF16(a1, b1, c[1][1]);
    c[1][2] = WMMA_BF16(a1, b2, c[1][2]);
    c[1][3] = WMMA_BF16(a1, b3, c[1][3]);
  }

  // Epilogue: C layout VGPR r -> M=r(+8 for lanes>=16), N=lane&15.
  // Adjacent lanes hold adjacent gate_up cols (gate on even, linear on odd).
  const int half = lane >> 4;
  const int nIdx = lane & 15;
#pragma unroll
  for (int ni = 0; ni < 4; ++ni) {
    const int col = cBase + ni * 16 + nIdx;
    const float bias = bgu[e * (2 * ND) + col];
#pragma unroll
    for (int mi = 0; mi < 2; ++mi) {
#pragma unroll
      for (int r = 0; r < 8; ++r) {
        float v = c[mi][ni][r] + bias;
        float other = __shfl_xor(v, 1, 32);   // partner column (col^1), same t
        float g = fminf(v, LIMIT_);
        float l = fminf(fmaxf(other, -LIMIT_), LIMIT_);
        float sig = 1.0f / (1.0f + __expf(-ALPHA_ * g));
        float f = g * sig * (l + 1.0f);
        if ((lane & 1) == 0) {                // even lane: v is the gate column
          const int t = tBase + mi * 16 + half * 8 + r;
          const int j = col >> 1;
          fused[((size_t)e * NT + t) * ND + j] = (__bf16)f;
        }
      }
    }
  }
}

// ---------------------------------------------------------------------------
// K2: down GEMM (bf16 WMMA) + bias + routing-weighted combine over all experts
// block = 256 thr, tile = 64 tokens x 128 hidden cols; grid = (H/128=8, T/64=32)
// Wave tile stays 32x32: needs both per-expert WMMA accs and persistent accs.
// ---------------------------------------------------------------------------
__global__ __launch_bounds__(256) void k_down_combine(
    const __bf16* __restrict__ fused, // [E,T,D]
    const __bf16* __restrict__ wd,    // [E,H,D]
    const float*  __restrict__ bd,    // [E,H]
    const float*  __restrict__ rw,    // [T,E]
    float* __restrict__ out)          // [T,H]
{
  __shared__ float wlds[64][NE];

  const int tid  = threadIdx.x;
  const int tBlk = blockIdx.y * 64;
  for (int i = tid; i < 64 * NE; i += 256)
    wlds[i >> 4][i & 15] = rw[(size_t)(tBlk + (i >> 4)) * NE + (i & 15)];
  __syncthreads();

  const int lane = tid & 31;
  const int wave = tid >> 5;
  const int wm = wave >> 2, wn = wave & 3;
  const int hBase = blockIdx.x * 128 + wn * 32;
  const int tBase = tBlk + wm * 32;
  const int half = lane >> 4;
  const int nIdx = lane & 15;

  f32x8 acc[2][2] = {};   // persistent routing-weighted accumulators

  for (int e = 0; e < NE; ++e) {
    const __bf16* fa = fused + ((size_t)e * NT + tBase) * ND;
    const __bf16* wb = wd + ((size_t)e * NH + hBase) * ND;
    const __bf16* fa1 = fa + (size_t)16 * ND;
    const __bf16* wb1 = wb + (size_t)16 * ND;

    f32x8 c[2][2] = {};
    for (int kk = 0; kk < ND; kk += 32) {
      bf16x16 a0 = load_a_frag(fa,  ND, kk, lane);
      bf16x16 a1 = load_a_frag(fa1, ND, kk, lane);
      bf16x16 b0 = load_b_frag(wb,  ND, kk, lane);
      bf16x16 b1 = load_b_frag(wb1, ND, kk, lane);
      __builtin_prefetch((const void*)(wb + (size_t)(lane & 15) * ND + kk + 32), 0, 1);
      c[0][0] = WMMA_BF16(a0, b0, c[0][0]);
      c[0][1] = WMMA_BF16(a0, b1, c[0][1]);
      c[1][0] = WMMA_BF16(a1, b0, c[1][0]);
      c[1][1] = WMMA_BF16(a1, b1, c[1][1]);
    }

#pragma unroll
    for (int ni = 0; ni < 2; ++ni) {
      const int h = hBase + ni * 16 + nIdx;
      const float bde = bd[e * NH + h];
#pragma unroll
      for (int mi = 0; mi < 2; ++mi) {
#pragma unroll
        for (int r = 0; r < 8; ++r) {
          const int lt = wm * 32 + mi * 16 + half * 8 + r;  // token within block tile
          const float w = wlds[lt][e];
          acc[mi][ni][r] += w * (c[mi][ni][r] + bde);
        }
      }
    }
  }

#pragma unroll
  for (int ni = 0; ni < 2; ++ni) {
    const int h = hBase + ni * 16 + nIdx;
#pragma unroll
    for (int mi = 0; mi < 2; ++mi) {
#pragma unroll
      for (int r = 0; r < 8; ++r) {
        const int t = tBase + mi * 16 + half * 8 + r;
        out[(size_t)t * NH + h] = acc[mi][ni][r];
      }
    }
  }
}

// ---------------------------------------------------------------------------
extern "C" void kernel_launch(void* const* d_in, const int* in_sizes, int n_in,
                              void* d_out, int out_size, void* d_ws, size_t ws_size,
                              hipStream_t stream) {
  const float* x   = (const float*)d_in[0];  // [B,S,H] -> [T,H]
  const float* rw  = (const float*)d_in[1];  // [T,E]
  // d_in[2] router_indices: unused by the dense reference
  const float* Wgu = (const float*)d_in[3];  // [E,2D,H]
  const float* bgu = (const float*)d_in[4];  // [E,2D]
  const float* Wd  = (const float*)d_in[5];  // [E,H,D]
  const float* bd  = (const float*)d_in[6];  // [E,H]
  float* out = (float*)d_out;                // [T,H]

  char* ws = (char*)d_ws;
  size_t off = 0;
  __bf16* xb   = (__bf16*)(ws + off); off += (size_t)NT * NH * 2;            //  4 MB
  __bf16* wgub = (__bf16*)(ws + off); off += (size_t)NE * 2 * ND * NH * 2;   // 64 MB
  __bf16* wdb  = (__bf16*)(ws + off); off += (size_t)NE * NH * ND * 2;       // 32 MB
  __bf16* fus  = (__bf16*)(ws + off); off += (size_t)NE * NT * ND * 2;       // 64 MB

  // K0: casts (every region fully written before it is read)
  k_cast_bf16<<<2048, 256, 0, stream>>>(x,   xb,   NT * NH);
  k_cast_bf16<<<4096, 256, 0, stream>>>(Wgu, wgub, NE * 2 * ND * NH);
  k_cast_bf16<<<4096, 256, 0, stream>>>(Wd,  wdb,  NE * NH * ND);

  // K1: gate_up + swiglu
  k_gateup_swiglu<<<dim3(2 * ND / 256, NT / 64, NE), 256, 0, stream>>>(xb, wgub, bgu, fus);

  // K2: down projection + expert combine
  k_down_combine<<<dim3(NH / 128, NT / 64), 256, 0, stream>>>(fus, wdb, bd, rw, out);
}
